// VideoTransformer_63402307224452
// MI455X (gfx1250) — matrix operand order repaired and checked by reading, not compiled, not run on hardware
//
#include <hip/hip_runtime.h>
#include <math.h>

// ---------------------------------------------------------------------------
// Types for CDNA5 WMMA (wave32): v_wmma_f32_16x16x32_bf16
// ---------------------------------------------------------------------------
typedef __attribute__((ext_vector_type(16))) __bf16 v16bf;
typedef __attribute__((ext_vector_type(8)))  float  v8f;

union FragU {
    v16bf v;
    uint4 q[2];
    unsigned short u[16];
};

__device__ __forceinline__ unsigned short f2bf(float f) {
    unsigned u = __float_as_uint(f);
    unsigned r = u + 0x7FFFu + ((u >> 16) & 1u);   // round-to-nearest-even
    return (unsigned short)(r >> 16);
}

// A-matrix fragment (16x32 bf16): lane holds row = (lane&15);
// elems j -> K = kb+j (contiguous), kb = (lane>>4)*16.
__device__ __forceinline__ v16bf load_frag_a(const unsigned short* s, int row, int ld, int lane) {
    int kb = (lane >> 4) * 16;
    FragU f;
    f.q[0] = *(const uint4*)(s + row * ld + kb);
    f.q[1] = *(const uint4*)(s + row * ld + kb + 8);
    return f.v;
}

// B-matrix fragment (32x16 bf16): lane holds col = (lane&15);
// elems e -> K = (lane>>4)*16 + e (contiguous). Source stored [col][K].
__device__ __forceinline__ v16bf load_frag_b(const unsigned short* s, int row, int ld, int lane) {
    int kb = (lane >> 4) * 16;
    FragU f;
    f.q[0] = *(const uint4*)(s + row * ld + kb);
    f.q[1] = *(const uint4*)(s + row * ld + kb + 8);
    return f.v;
}

// ---------------------------------------------------------------------------
// Tensor Data Mover: 2D tile (32 bf16 x 128 rows) global -> LDS, with LDS
// row padding 16B every 64B so rows land at stride 40 elems (conflict-free).
// Descriptor packing per CDNA5 ISA ch.8 (group0 / group1).
// ---------------------------------------------------------------------------
#if defined(__HIP_DEVICE_COMPILE__) && defined(__has_builtin)
#if __has_builtin(__builtin_amdgcn_tensor_load_to_lds)
#define USE_TDM 1
#endif
#endif
#ifndef USE_TDM
#define USE_TDM 0
#endif

#if USE_TDM
typedef unsigned int u32x4 __attribute__((ext_vector_type(4)));
typedef int          i32x4 __attribute__((ext_vector_type(4)));
typedef int          i32x8 __attribute__((ext_vector_type(8)));

__device__ __forceinline__ void tdm_load_tile32x128(
    unsigned lds_addr, const void* gptr, unsigned stride_elems)
{
    unsigned long long ga = (unsigned long long)gptr;
    u32x4 g0;
    g0[0] = 1u;                                            // count=1 (valid user D#)
    g0[1] = lds_addr;                                      // LDS byte address
    g0[2] = (unsigned)(ga & 0xFFFFFFFFu);                  // global_addr[31:0]
    g0[3] = (unsigned)((ga >> 32) & 0x01FFFFFFu) | (2u << 30); // addr[56:32] | type=2
    i32x8 g1;
    // data_size=2B (1<<16), pad_enable (1<<20), pad_interval=16dw (3<<22),
    // pad_amount=4dw (3<<25)
    g1[0] = (int)((1u << 16) | (1u << 20) | (3u << 22) | (3u << 25));
    g1[1] = 0;                                             // atomic addr, dim0[15:0]=0
    g1[2] = 16;                                            // tensor_dim0 = 1<<20 (hi16)
    g1[3] = (int)(16u | (32u << 16));                      // tensor_dim1=1<<20, tile_dim0=32
    g1[4] = 128;                                           // tile_dim1=128, tile_dim2=0
    g1[5] = (int)stride_elems;                             // tensor_dim0_stride[31:0]
    g1[6] = 0;                                             // stride hi / dim1_stride lo
    g1[7] = 0;
    i32x4 z4 = {0, 0, 0, 0};
#if __clang_major__ >= 23
    i32x8 z8 = {0, 0, 0, 0, 0, 0, 0, 0};
    __builtin_amdgcn_tensor_load_to_lds(g0, g1, z4, z4, z8, 0);
#else
    __builtin_amdgcn_tensor_load_to_lds(g0, g1, z4, z4, 0);
#endif
}
#endif

// ---------------------------------------------------------------------------
// copy fp32
// ---------------------------------------------------------------------------
__global__ void copy_kernel(const float* __restrict__ src, float* __restrict__ dst, int n) {
    int i = blockIdx.x * 256 + threadIdx.x;
    if (i < n) dst[i] = src[i];
}

// ---------------------------------------------------------------------------
// LayerNorm over C=768; writes bf16 (yb) or fp32 (yf)
// ---------------------------------------------------------------------------
__global__ __launch_bounds__(256) void ln_kernel(
    const float* __restrict__ x, const float* __restrict__ sc, const float* __restrict__ bi,
    unsigned short* __restrict__ yb, float* __restrict__ yf)
{
    constexpr int C = 768;
    __shared__ float ps[8], ps2[8], mb[2];
    const int rowi = blockIdx.x;
    const float* xr = x + (size_t)rowi * C;
    const int tid = threadIdx.x;
    float v0 = xr[tid], v1 = xr[tid + 256], v2 = xr[tid + 512];
    float s  = v0 + v1 + v2;
    float s2 = v0 * v0 + v1 * v1 + v2 * v2;
    #pragma unroll
    for (int off = 1; off < 32; off <<= 1) {
        s  += __shfl_xor(s,  off, 32);
        s2 += __shfl_xor(s2, off, 32);
    }
    int wave = tid >> 5, lane = tid & 31;
    if (lane == 0) { ps[wave] = s; ps2[wave] = s2; }
    __syncthreads();
    if (tid == 0) {
        float ts = 0.f, ts2 = 0.f;
        #pragma unroll
        for (int i = 0; i < 8; ++i) { ts += ps[i]; ts2 += ps2[i]; }
        float mu  = ts / C;
        float var = ts2 / C - mu * mu;
        mb[0] = mu;
        mb[1] = rsqrtf(var + 1e-6f);
    }
    __syncthreads();
    float mu = mb[0], rs = mb[1];
    #pragma unroll
    for (int i = 0; i < 3; ++i) {
        int c = tid + i * 256;
        float val = (xr[c] - mu) * rs * sc[c] + bi[c];
        if (yf) yf[(size_t)rowi * C + c] = val;
        else    yb[(size_t)rowi * C + c] = f2bf(val);
    }
}

// ---------------------------------------------------------------------------
// Transpose + fp32->bf16: W[K][N] -> Wt[N][K]
// ---------------------------------------------------------------------------
__global__ __launch_bounds__(256) void transpose_cvt(
    const float* __restrict__ W, unsigned short* __restrict__ Wt, int K, int N)
{
    __shared__ float tile[32][33];
    int k0 = blockIdx.x * 32;
    int n0 = blockIdx.y * 32;
    int tx = threadIdx.x & 31;
    int ty = threadIdx.x >> 5;
    #pragma unroll
    for (int i = 0; i < 4; ++i)
        tile[ty + i * 8][tx] = W[(size_t)(k0 + ty + i * 8) * N + n0 + tx];
    __syncthreads();
    #pragma unroll
    for (int i = 0; i < 4; ++i)
        Wt[(size_t)(n0 + ty + i * 8) * K + k0 + tx] = f2bf(tile[tx][ty + i * 8]);
}

// ---------------------------------------------------------------------------
// bf16 WMMA GEMM:  out = A[MxK] @ Wt[NxK]^T + bias
//   mode 0: store bf16   mode 1: GELU(exact)+bf16   mode 2: resid += (fp32)
// Block 256 (8 waves), tile 128x128x32; wave computes 64x32 (4x2 accums).
// TDM path: double-buffered LDS, wave 0 drives tensor_load_to_lds pipeline.
// ---------------------------------------------------------------------------
#define GLDS 40   // LDS row stride in bf16 elems (80B: 16B aligned, conflict-free)

__global__ __launch_bounds__(256) void gemm_bf16(
    const unsigned short* __restrict__ A,
    const unsigned short* __restrict__ Wt,
    const float* __restrict__ bias,
    unsigned short* __restrict__ outb,
    float* __restrict__ resid,
    int M, int N, int K, int mode)
{
    (void)M;
#if USE_TDM
    __shared__ unsigned short smem[2 * 2 * 128 * GLDS];   // [buf][A|B][128*40]
#else
    __shared__ unsigned short smem[2 * 128 * GLDS];       // [A|B][128*40]
#endif
    const int tid  = threadIdx.x;
    const int lane = tid & 31;
    const int wave = tid >> 5;
    const int wm = (wave & 1) * 64;
    const int wn = (wave >> 1) * 32;
    const int m0 = blockIdx.x * 128;
    const int n0 = blockIdx.y * 128;
    const int tileElems = 128 * GLDS;

    v8f zero = {};
    v8f acc[4][2];
    #pragma unroll
    for (int i = 0; i < 4; ++i)
        #pragma unroll
        for (int j = 0; j < 2; ++j) acc[i][j] = zero;

    const int nIter = K / 32;

#if USE_TDM
    if (wave == 0) {
        tdm_load_tile32x128((unsigned)(size_t)smem,
                            A + (size_t)m0 * K, (unsigned)K);
        tdm_load_tile32x128((unsigned)(size_t)(smem + tileElems),
                            Wt + (size_t)n0 * K, (unsigned)K);
    }
    for (int kt = 0; kt < nIter; ++kt) {
        const unsigned short* sA = smem + (kt & 1) * 2 * tileElems;
        const unsigned short* sB = sA + tileElems;
        __syncthreads();                       // readers done with buf being refilled
        if (wave == 0) {
            if (kt + 1 < nIter) {
                unsigned short* dA = smem + ((kt + 1) & 1) * 2 * tileElems;
                tdm_load_tile32x128((unsigned)(size_t)dA,
                                    A + (size_t)m0 * K + (kt + 1) * 32, (unsigned)K);
                tdm_load_tile32x128((unsigned)(size_t)(dA + tileElems),
                                    Wt + (size_t)n0 * K + (kt + 1) * 32, (unsigned)K);
                __builtin_amdgcn_s_wait_tensorcnt(2);   // current buf's 2 DMAs done
            } else {
                __builtin_amdgcn_s_wait_tensorcnt(0);
            }
        }
        __syncthreads();                       // data visible to all waves
#else
    unsigned short* sA = smem;
    unsigned short* sB = smem + tileElems;
    const int row = tid >> 1;          // 0..127
    const int kh  = (tid & 1) * 16;    // 0 or 16
    for (int kt = 0; kt < nIter; ++kt) {
        __syncthreads();
        const unsigned short* ga = A  + (size_t)(m0 + row) * K + kt * 32 + kh;
        const unsigned short* gb = Wt + (size_t)(n0 + row) * K + kt * 32 + kh;
        uint4 a0 = *(const uint4*)ga;
        uint4 a1 = *(const uint4*)(ga + 8);
        uint4 b0 = *(const uint4*)gb;
        uint4 b1 = *(const uint4*)(gb + 8);
        *(uint4*)(sA + row * GLDS + kh)     = a0;
        *(uint4*)(sA + row * GLDS + kh + 8) = a1;
        *(uint4*)(sB + row * GLDS + kh)     = b0;
        *(uint4*)(sB + row * GLDS + kh + 8) = b1;
        if (kt + 1 < nIter) {
            __builtin_prefetch(ga + 32, 0, 1);
            __builtin_prefetch(gb + 32, 0, 1);
        }
        __syncthreads();
#endif
        v16bf bF[2];
        #pragma unroll
        for (int ni = 0; ni < 2; ++ni)
            bF[ni] = load_frag_b(sB, wn + ni * 16 + (lane & 15), GLDS, lane);
        #pragma unroll
        for (int mi = 0; mi < 4; ++mi) {
            v16bf aF = load_frag_a(sA, wm + mi * 16 + (lane & 15), GLDS, lane);
            #pragma unroll
            for (int ni = 0; ni < 2; ++ni)
                acc[mi][ni] = __builtin_amdgcn_wmma_f32_16x16x32_bf16(
                    false, aF, false, bF[ni], (short)0, acc[mi][ni], false, false);
        }
    }

    // Epilogue. C/D layout: vreg v -> row = v + 8*(lane>>4), col = lane&15.
    const int colH = lane & 15;
    const int rowH = (lane >> 4) * 8;
    #pragma unroll
    for (int mi = 0; mi < 4; ++mi) {
        #pragma unroll
        for (int ni = 0; ni < 2; ++ni) {
            int n = n0 + wn + ni * 16 + colH;
            float bv = bias[n];
            #pragma unroll
            for (int v = 0; v < 8; ++v) {
                int m = m0 + wm + mi * 16 + rowH + v;
                float val = acc[mi][ni][v] + bv;
                if (mode == 2) {
                    resid[(size_t)m * N + n] += val;
                } else {
                    if (mode == 1)
                        val = 0.5f * val * (1.0f + erff(val * 0.70710678118654752f));
                    outb[(size_t)m * N + n] = f2bf(val);
                }
            }
        }
    }
}

// ---------------------------------------------------------------------------
// Flash attention, bf16 WMMA, online softmax.
// qkv layout: (B, N, 3, H, 64) bf16. grid (16 qblocks, 48 b*h), block 128.
// ---------------------------------------------------------------------------
__global__ __launch_bounds__(128) void attn_kernel(
    const unsigned short* __restrict__ qkv,
    const int* __restrict__ mask,
    unsigned short* __restrict__ ao)
{
    constexpr int Nn = 1024, Hh = 12, Dd = 64;
    constexpr float SCALE = 0.125f;  // 64^-0.5
    __shared__ unsigned short sK[32 * 72];       // [key][d], stride 72 (144B)
    __shared__ unsigned short sV[64 * GLDS];     // [d][key], stride 40 (80B)
    __shared__ unsigned short sP[4 * 16 * GLDS]; // per-wave P [q][key]

    const int tid  = threadIdx.x;
    const int lane = tid & 31;
    const int wave = tid >> 5;
    const int bh = blockIdx.y;
    const int b  = bh / Hh, hh = bh % Hh;
    const int qbase = blockIdx.x * 64 + wave * 16;
    const int kbA = (lane >> 4) * 8;

    // Preload Q fragments (16x64 = two 16x32 A-frags) straight from global.
    const int qrow = qbase + (lane & 15);
    const unsigned short* qptr = qkv + ((size_t)((b * Nn + qrow) * 3 + 0) * Hh + hh) * Dd;
    v16bf aQ[2];
    #pragma unroll
    for (int ks = 0; ks < 2; ++ks) {
        FragU f;
        f.q[0] = *(const uint4*)(qptr + ks * 32 + kbA * 2);
        f.q[1] = *(const uint4*)(qptr + ks * 32 + kbA * 2 + 8);
        aQ[ks] = f.v;
    }

    v8f zero = {};
    float mrow[8], lrow[8];
    v8f o[4];
    #pragma unroll
    for (int v = 0; v < 8; ++v) { mrow[v] = -1e30f; lrow[v] = 0.f; }
    #pragma unroll
    for (int dt = 0; dt < 4; ++dt) o[dt] = zero;

    unsigned short* sPw = sP + wave * 16 * GLDS;
    const int key_l = tid >> 2;           // 0..31
    const int d0    = (tid & 3) * 16;     // 0,16,32,48

    for (int kb = 0; kb < Nn; kb += 32) {
        __syncthreads();
        {   // stage K (row-major) and V (transposed) for this 32-key chunk
            const unsigned short* kp = qkv + ((size_t)((b * Nn + kb + key_l) * 3 + 1) * Hh + hh) * Dd + d0;
            uint4 k0 = *(const uint4*)kp;
            uint4 k1 = *(const uint4*)(kp + 8);
            *(uint4*)(sK + key_l * 72 + d0)     = k0;
            *(uint4*)(sK + key_l * 72 + d0 + 8) = k1;
            const unsigned short* vp = qkv + ((size_t)((b * Nn + kb + key_l) * 3 + 2) * Hh + hh) * Dd + d0;
            FragU vt;
            vt.q[0] = *(const uint4*)vp;
            vt.q[1] = *(const uint4*)(vp + 8);
            #pragma unroll
            for (int i = 0; i < 16; ++i)
                sV[(d0 + i) * GLDS + key_l] = vt.u[i];
        }
        __syncthreads();

        // S = Q @ K^T : two 16x16 key tiles, contraction over d (2 k-steps)
        v8f s[2];
        #pragma unroll
        for (int t = 0; t < 2; ++t) {
            s[t] = zero;
            int krow = t * 16 + (lane & 15);
            #pragma unroll
            for (int ks = 0; ks < 2; ++ks) {
                FragU f;
                f.q[0] = *(const uint4*)(sK + krow * 72 + ks * 32 + (lane >> 4) * 16);
                f.q[1] = *(const uint4*)(sK + krow * 72 + ks * 32 + (lane >> 4) * 16 + 8);
                s[t] = __builtin_amdgcn_wmma_f32_16x16x32_bf16(
                    false, aQ[ks], false, f.v, (short)0, s[t], false, false);
            }
        }

        // scale + mask bias (finite -1e4, same as reference)
        float bias0 = (mask[b * Nn + kb + (lane & 15)]      == 0) ? -10000.f : 0.f;
        float bias1 = (mask[b * Nn + kb + 16 + (lane & 15)] == 0) ? -10000.f : 0.f;
        float tmax[8];
        #pragma unroll
        for (int v = 0; v < 8; ++v) {
            float x0 = s[0][v] * SCALE + bias0;
            float x1 = s[1][v] * SCALE + bias1;
            s[0][v] = x0; s[1][v] = x1;
            tmax[v] = fmaxf(x0, x1);
        }

        // online softmax: rows live per half-wave; reduce across 16 lanes
        float alpha[8];
        #pragma unroll
        for (int v = 0; v < 8; ++v) {
            float mx = tmax[v];
            mx = fmaxf(mx, __shfl_xor(mx, 1, 16));
            mx = fmaxf(mx, __shfl_xor(mx, 2, 16));
            mx = fmaxf(mx, __shfl_xor(mx, 4, 16));
            mx = fmaxf(mx, __shfl_xor(mx, 8, 16));
            float mn = fmaxf(mrow[v], mx);
            alpha[v] = __expf(mrow[v] - mn);
            mrow[v] = mn;
            float p0 = __expf(s[0][v] - mn);
            float p1 = __expf(s[1][v] - mn);
            s[0][v] = p0; s[1][v] = p1;
            float rs = p0 + p1;
            rs += __shfl_xor(rs, 1, 16);
            rs += __shfl_xor(rs, 2, 16);
            rs += __shfl_xor(rs, 4, 16);
            rs += __shfl_xor(rs, 8, 16);
            lrow[v] = lrow[v] * alpha[v] + rs;
        }
        #pragma unroll
        for (int dt = 0; dt < 4; ++dt)
            #pragma unroll
            for (int v = 0; v < 8; ++v)
                o[dt][v] *= alpha[v];

        // P (D-layout) -> LDS -> A-layout fragment
        const int qr = (lane >> 4) * 8;
        #pragma unroll
        for (int t = 0; t < 2; ++t)
            #pragma unroll
            for (int v = 0; v < 8; ++v)
                sPw[(qr + v) * GLDS + t * 16 + (lane & 15)] = f2bf(s[t][v]);

        FragU pf;
        int q = lane & 15;
        pf.q[0] = *(const uint4*)(sPw + q * GLDS + kbA * 2);
        pf.q[1] = *(const uint4*)(sPw + q * GLDS + kbA * 2 + 8);

        // O += P @ V (contraction over 32 keys)
        #pragma unroll
        for (int dt = 0; dt < 4; ++dt) {
            int drow = dt * 16 + (lane & 15);
            FragU vf;
            vf.q[0] = *(const uint4*)(sV + drow * GLDS + (lane >> 4) * 16);
            vf.q[1] = *(const uint4*)(sV + drow * GLDS + (lane >> 4) * 16 + 8);
            o[dt] = __builtin_amdgcn_wmma_f32_16x16x32_bf16(
                false, pf.v, false, vf.v, (short)0, o[dt], false, false);
        }
    }

    // normalize and write (B, N, C) bf16, head hh occupies cols [hh*64, hh*64+64)
    #pragma unroll
    for (int dt = 0; dt < 4; ++dt) {
        int d = dt * 16 + (lane & 15);
        #pragma unroll
        for (int v = 0; v < 8; ++v) {
            int q = qbase + (lane >> 4) * 8 + v;
            float val = o[dt][v] / lrow[v];
            ao[(size_t)(b * Nn + q) * 768 + hh * 64 + d] = f2bf(val);
        }
    }
}

// ---------------------------------------------------------------------------
// Host orchestration
// ---------------------------------------------------------------------------
extern "C" void kernel_launch(void* const* d_in, const int* in_sizes, int n_in,
                              void* d_out, int out_size, void* d_ws, size_t ws_size,
                              hipStream_t stream) {
    (void)in_sizes; (void)n_in; (void)out_size; (void)ws_size;
    const float* x      = (const float*)d_in[0];
    const int*   vmask  = (const int*)  d_in[1];
    const float* qkv_w  = (const float*)d_in[2];
    const float* qkv_b  = (const float*)d_in[3];
    const float* proj_w = (const float*)d_in[4];
    const float* proj_b = (const float*)d_in[5];
    const float* n1_s   = (const float*)d_in[6];
    const float* n1_b   = (const float*)d_in[7];
    const float* n2_s   = (const float*)d_in[8];
    const float* n2_b   = (const float*)d_in[9];
    const float* fc1_w  = (const float*)d_in[10];
    const float* fc1_b  = (const float*)d_in[11];
    const float* fc2_w  = (const float*)d_in[12];
    const float* fc2_b  = (const float*)d_in[13];
    const float* fn_s   = (const float*)d_in[14];
    const float* fn_b   = (const float*)d_in[15];
    float* out = (float*)d_out;

    const int BN_ = 4 * 1024;     // 4096 rows
    const int C   = 768, C3 = 2304, HID = 3072;

    char* wsb = (char*)d_ws;
    size_t off = 0;
    auto carve = [&](size_t bytes) -> void* {
        void* p = wsb + off;
        off += (bytes + 255) & ~(size_t)255;
        return p;
    };
    float*          h    = (float*)         carve((size_t)BN_ * C * 4);
    unsigned short* y    = (unsigned short*)carve((size_t)BN_ * C * 2);
    unsigned short* qkvb = (unsigned short*)carve((size_t)BN_ * C3 * 2);
    unsigned short* aob  = (unsigned short*)carve((size_t)BN_ * C * 2);
    unsigned short* hid  = (unsigned short*)carve((size_t)BN_ * HID * 2);
    unsigned short* wtQ  = (unsigned short*)carve((size_t)C3 * C * 2);
    unsigned short* wtP  = (unsigned short*)carve((size_t)C * C * 2);
    unsigned short* wtF1 = (unsigned short*)carve((size_t)HID * C * 2);
    unsigned short* wtF2 = (unsigned short*)carve((size_t)C * HID * 2);

    // h = x
    copy_kernel<<<(BN_ * C + 255) / 256, 256, 0, stream>>>(x, h, BN_ * C);

    for (int l = 0; l < 12; ++l) {
        ln_kernel<<<BN_, 256, 0, stream>>>(h, n1_s + l * C, n1_b + l * C, y, nullptr);
        transpose_cvt<<<dim3(C / 32, C3 / 32), 256, 0, stream>>>(qkv_w + (size_t)l * C * C3, wtQ, C, C3);
        gemm_bf16<<<dim3(BN_ / 128, C3 / 128), 256, 0, stream>>>(y, wtQ, qkv_b + (size_t)l * C3,
                                                                 qkvb, nullptr, BN_, C3, C, 0);
        attn_kernel<<<dim3(16, 48), 128, 0, stream>>>(qkvb, vmask, aob);
        transpose_cvt<<<dim3(C / 32, C / 32), 256, 0, stream>>>(proj_w + (size_t)l * C * C, wtP, C, C);
        gemm_bf16<<<dim3(BN_ / 128, C / 128), 256, 0, stream>>>(aob, wtP, proj_b + (size_t)l * C,
                                                                nullptr, h, BN_, C, C, 2);
        ln_kernel<<<BN_, 256, 0, stream>>>(h, n2_s + l * C, n2_b + l * C, y, nullptr);
        transpose_cvt<<<dim3(C / 32, HID / 32), 256, 0, stream>>>(fc1_w + (size_t)l * C * HID, wtF1, C, HID);
        gemm_bf16<<<dim3(BN_ / 128, HID / 128), 256, 0, stream>>>(y, wtF1, fc1_b + (size_t)l * HID,
                                                                  hid, nullptr, BN_, HID, C, 1);
        transpose_cvt<<<dim3(HID / 32, C / 32), 256, 0, stream>>>(fc2_w + (size_t)l * HID * C, wtF2, HID, C);
        gemm_bf16<<<dim3(BN_ / 128, C / 128), 256, 0, stream>>>(hid, wtF2, fc2_b + (size_t)l * C,
                                                                nullptr, h, BN_, C, HID, 2);
    }
    ln_kernel<<<BN_, 256, 0, stream>>>(h, fn_s, fn_b, nullptr, out);
}
